// MHA_69552700391933
// MI455X (gfx1250) — matrix operand order repaired and checked by reading, compiled, not run on hardware
//
#include <hip/hip_runtime.h>
#include <hip/hip_bf16.h>
#include <stdint.h>

typedef __bf16 bf16_t;
typedef __attribute__((ext_vector_type(16))) bf16_t v16bf;
typedef __attribute__((ext_vector_type(8)))  float  v8f;

union ABFrag { v16bf v; uint32_t u[8]; };

__device__ inline uint32_t pack_bf2(float a, float b) {
  uint16_t lo = __builtin_bit_cast(uint16_t, (bf16_t)a);
  uint16_t hi = __builtin_bit_cast(uint16_t, (bf16_t)b);
  return (uint32_t)lo | ((uint32_t)hi << 16);
}

__device__ inline v8f wmma_bf16(const ABFrag& a, const ABFrag& b, v8f c) {
  // D = A(16x32 bf16) x B(32x16 bf16) + C(16x16 f32)
  return __builtin_amdgcn_wmma_f32_16x16x32_bf16(
      /*neg_a=*/false, a.v, /*neg_b=*/false, b.v,
      /*c_mod=*/(short)0, c, /*reuse_a=*/false, /*reuse_b=*/false);
}

// ---- CDNA5 async global->LDS copy (ASYNCcnt-tracked, bypasses VGPRs) ------
// Low 32 bits of a generic pointer into LDS are the LDS byte address
// (ISA 10.2, LDS aperture).
__device__ inline void async_copy16(void* lds_ptr, const void* gptr) {
  uint32_t lds_addr   = (uint32_t)(uintptr_t)lds_ptr;
  unsigned long long ga = (unsigned long long)(uintptr_t)gptr;
  asm volatile("global_load_async_to_lds_b128 %0, %1, off"
               :: "v"(lds_addr), "v"(ga) : "memory");
}

__device__ inline void async_wait0() {
#if __has_builtin(__builtin_amdgcn_s_wait_asynccnt)
  __builtin_amdgcn_s_wait_asynccnt(0);
#else
  asm volatile("s_wait_asynccnt 0x0" ::: "memory");
#endif
}

// ---------------------------------------------------------------------------
// C[M x N] = A[M x K] @ B[N x K]^T   (all row-major; B always f32 in global)
// A source is f32 or bf16 (template). Output f32 or bf16 (template).
// Block: 256 threads = 8 waves; 128x128 tile; BK=32; waves 2(M) x 4(N),
// each wave computes 64x32 = 4x2 WMMA tiles.
// bf16 A tiles are staged with GLOBAL_LOAD_ASYNC_TO_LDS_B128.
// ---------------------------------------------------------------------------
template <bool A_BF16, bool OUT_BF16>
__global__ __launch_bounds__(256)
void gemm_bt_kernel(const void* __restrict__ Ap, const float* __restrict__ B,
                    void* __restrict__ Cp, int N, int K)
{
  constexpr int BM = 128, BN = 128, BK = 32;
  __shared__ bf16_t As[BM * BK];
  __shared__ bf16_t Bs[BN * BK];

  const int tid  = threadIdx.x;
  const int lane = tid & 31;
  const int half = lane >> 4;     // 0 = lanes 0-15, 1 = lanes 16-31
  const int lp   = lane & 15;
  const int wave = tid >> 5;
  const int wm   = wave >> 2;     // 0..1
  const int wn   = wave & 3;      // 0..3
  const int m0   = blockIdx.y * BM;
  const int n0   = blockIdx.x * BN;

  const int ldrow = tid >> 1;          // 0..127
  const int ldc   = (tid & 1) * 16;    // 0 or 16

  const v8f vzero = {0.f,0.f,0.f,0.f,0.f,0.f,0.f,0.f};
  v8f acc[4][2];
  #pragma unroll
  for (int i = 0; i < 4; ++i)
    #pragma unroll
    for (int j = 0; j < 2; ++j) acc[i][j] = vzero;

  uint32_t* Asu = (uint32_t*)As;
  uint32_t* Bsu = (uint32_t*)Bs;

  for (int kt = 0; kt < K; kt += BK) {
    // ---- stage A tile ----
    if (A_BF16) {
      const bf16_t* s = (const bf16_t*)Ap + (size_t)(m0 + ldrow) * K + kt + ldc;
      bf16_t* d = As + ldrow * BK + ldc;
      async_copy16(d,     s);        // 16 bf16 = 2 x b128, LDS-direct
      async_copy16(d + 8, s + 8);
    } else {
      const float* s = (const float*)Ap + (size_t)(m0 + ldrow) * K + kt + ldc;
      uint32_t* d = Asu + (ldrow * BK + ldc) / 2;
      #pragma unroll
      for (int i = 0; i < 8; ++i) d[i] = pack_bf2(s[2*i], s[2*i+1]);
    }
    // ---- stage B tile (f32 -> bf16) ----
    {
      const float* s = B + (size_t)(n0 + ldrow) * K + kt + ldc;
      uint32_t* d = Bsu + (ldrow * BK + ldc) / 2;
      #pragma unroll
      for (int i = 0; i < 8; ++i) d[i] = pack_bf2(s[2*i], s[2*i+1]);
    }
    // ---- prefetch next K tile while we compute this one ----
    if (kt + BK < K) {
      const char* apre = A_BF16
          ? (const char*)((const bf16_t*)Ap + (size_t)(m0 + ldrow) * K + kt + BK + ldc)
          : (const char*)((const float*)Ap + (size_t)(m0 + ldrow) * K + kt + BK + ldc);
      __builtin_prefetch(apre, 0, 0);
      __builtin_prefetch(B + (size_t)(n0 + ldrow) * K + kt + BK + ldc, 0, 0);
    }
    if (A_BF16) async_wait0();       // drain ASYNCcnt before barrier
    __syncthreads();

    // ---- B fragments (K=32 x N=16, bf16): lane lp = col, half picks K 0-15/16-31
    ABFrag bfr[2];
    #pragma unroll
    for (int nt = 0; nt < 2; ++nt) {
      const int col = wn * 32 + nt * 16 + lp;
      #pragma unroll
      for (int j = 0; j < 8; ++j) {
        const int k = half * 16 + 2 * j;
        bfr[nt].u[j] = Bsu[(col * BK + k) >> 1];
      }
    }
    // ---- A fragments (M=16 x K=32, bf16) + WMMAs
    #pragma unroll
    for (int mt = 0; mt < 4; ++mt) {
      ABFrag afr;
      const int row = wm * 64 + mt * 16 + lp;
      #pragma unroll
      for (int j = 0; j < 8; ++j) {
        const int k = half * 16 + 2 * j;
        afr.u[j] = Asu[(row * BK + k) >> 1];
      }
      #pragma unroll
      for (int nt = 0; nt < 2; ++nt)
        acc[mt][nt] = wmma_bf16(afr, bfr[nt], acc[mt][nt]);
    }
    __syncthreads();
  }

  // ---- epilogue: C layout = lane lp -> col, element r -> row r + half*8
  #pragma unroll
  for (int mt = 0; mt < 4; ++mt)
    #pragma unroll
    for (int nt = 0; nt < 2; ++nt)
      #pragma unroll
      for (int r = 0; r < 8; ++r) {
        const int rr = m0 + wm * 64 + mt * 16 + half * 8 + r;
        const int cc = n0 + wn * 32 + nt * 16 + lp;
        if (OUT_BF16)
          ((bf16_t*)Cp)[(size_t)rr * N + cc] = (bf16_t)acc[mt][nt][r];
        else
          ((float*)Cp)[(size_t)rr * N + cc] = acc[mt][nt][r];
      }
}

// ---------------------------------------------------------------------------
// In-place RoPE on the Q (cols 0..4095) and K (cols 4096..5119) slices of the
// bf16 qkv buffer [2048 x 6144]. One thread per (s, even/odd pair).
// ---------------------------------------------------------------------------
__global__ __launch_bounds__(256)
void rope_kernel(bf16_t* __restrict__ qkv,
                 const float* __restrict__ fc, const float* __restrict__ fs)
{
  const int t = blockIdx.x * blockDim.x + threadIdx.x;   // 2048 * 2560 threads
  const int s = t / 2560;
  const int u = t % 2560;
  const int p = (u < 2048) ? u : (u - 2048);
  const int h = p >> 6;          // head (32 for Q, 8 for K)
  const int i = p & 63;          // frequency index
  const int off = ((u < 2048) ? 0 : 4096) + h * 128 + 2 * i;

  bf16_t* ptr = qkv + (size_t)s * 6144 + off;
  const float e  = (float)ptr[0];
  const float o  = (float)ptr[1];
  const float c  = fc[s * 64 + i];
  const float sn = fs[s * 64 + i];
  ptr[0] = (bf16_t)(e * c - o * sn);
  ptr[1] = (bf16_t)(e * sn + o * c);
}

// ---------------------------------------------------------------------------
// Flash attention. Grid: (16 q-blocks of 128 rows, 32 heads). 256 threads =
// 8 waves; each wave owns 16 query rows. K tile 64x128 staged via
// GLOBAL_LOAD_ASYNC_TO_LDS_B128; V tile stored transposed (d-major) so PV
// B-fragments are contiguous. Causal mask computed inline.
// ---------------------------------------------------------------------------
__global__ __launch_bounds__(256)
void attn_kernel(const bf16_t* __restrict__ qkvb, bf16_t* __restrict__ Ob)
{
  __shared__ bf16_t Ks[64 * 128];   // [key][d]
  __shared__ bf16_t Vt[128 * 64];   // [d][key]
  __shared__ bf16_t Ps[8 * 16 * 64];// per-wave P staging [wave][row][key]

  const int tid  = threadIdx.x;
  const int lane = tid & 31;
  const int half = lane >> 4;
  const int lp   = lane & 15;
  const int wave = tid >> 5;
  const int head = blockIdx.y;
  const int kvh  = head >> 2;       // grouped KV: 32 q heads -> 8 kv heads
  const int qbase = blockIdx.x * 128;
  const int qrow  = qbase + wave * 16 + lp;   // A-fragment row for this lane

  // ---- load Q fragments once (16 rows x 128 head-dim = 4 K-steps) ----
  ABFrag qf[4];
  {
    const bf16_t* qp = qkvb + (size_t)qrow * 6144 + head * 128;
    #pragma unroll
    for (int ks = 0; ks < 4; ++ks)
      #pragma unroll
      for (int j = 0; j < 8; ++j) {
        const int k = ks * 32 + half * 16 + 2 * j;
        qf[ks].u[j] = *(const uint32_t*)(qp + k);
      }
  }

  const v8f vzero = {0.f,0.f,0.f,0.f,0.f,0.f,0.f,0.f};
  v8f o[8];
  float m[8], l[8];
  #pragma unroll
  for (int dt = 0; dt < 8; ++dt) o[dt] = vzero;
  #pragma unroll
  for (int r = 0; r < 8; ++r) { m[r] = -3.0e38f; l[r] = 0.f; }

  uint32_t* vtu = (uint32_t*)Vt;
  const uint32_t* Ku = (const uint32_t*)Ks;
  const uint32_t* Pu = (const uint32_t*)Ps;
  const uint32_t* Vu = (const uint32_t*)Vt;

  const int nkb = 2 * (blockIdx.x + 1);     // causal: only key blocks <= diag
  for (int kb = 0; kb < nkb; ++kb) {
    const int kstart = kb * 64;

    // ---- stage K tile [64 x 128] via async LDS-direct loads ----
    {
      const int row = tid >> 2;
      const int cb  = (tid & 3) * 32;
      const bf16_t* s =
          qkvb + (size_t)(kstart + row) * 6144 + 4096 + kvh * 128 + cb;
      bf16_t* d = Ks + row * 128 + cb;
      #pragma unroll
      for (int i = 0; i < 4; ++i)
        async_copy16(d + i * 8, s + i * 8);   // 4 x 16B = 32 bf16
    }
    // ---- stage V tile transposed: Vt[d][k], pairs (k,k+1) packed ----
    {
      const int d0  = tid >> 1;            // 0..127
      const int kb0 = (tid & 1) * 32;
      #pragma unroll
      for (int kk = 0; kk < 16; ++kk) {
        const int k = kb0 + 2 * kk;
        const bf16_t* vp =
            qkvb + (size_t)(kstart + k) * 6144 + 5120 + kvh * 128 + d0;
        const uint32_t a = (uint32_t)__builtin_bit_cast(uint16_t, vp[0]);
        const uint32_t b = (uint32_t)__builtin_bit_cast(uint16_t, vp[6144]);
        vtu[(d0 * 64 + k) >> 1] = a | (b << 16);
      }
    }
    async_wait0();                 // K-tile async writes visible in LDS
    __syncthreads();

    // ---- S = Q @ K^T : 4 key tiles x 4 K-steps = 16 WMMAs ----
    v8f s4[4];
    #pragma unroll
    for (int nt = 0; nt < 4; ++nt) s4[nt] = vzero;
    #pragma unroll
    for (int nt = 0; nt < 4; ++nt) {
      const int col = nt * 16 + lp;        // key index within tile
      #pragma unroll
      for (int ks = 0; ks < 4; ++ks) {
        ABFrag bf;
        #pragma unroll
        for (int j = 0; j < 8; ++j) {
          const int k = ks * 32 + half * 16 + 2 * j;   // head-dim index
          bf.u[j] = Ku[(col * 128 + k) >> 1];
        }
        s4[nt] = wmma_bf16(qf[ks], bf, s4[nt]);
      }
    }

    // ---- streaming softmax update ----
    const float scale = 0.08838834764831845f;   // 1/sqrt(128)
    #pragma unroll
    for (int r = 0; r < 8; ++r) {
      const int qr = qbase + wave * 16 + half * 8 + r;
      float vmax = -3.0e38f;
      #pragma unroll
      for (int nt = 0; nt < 4; ++nt) {
        float v = s4[nt][r] * scale;
        const int kc = kstart + nt * 16 + lp;
        if (kc > qr) v = -1.0e9f;
        s4[nt][r] = v;
        vmax = fmaxf(vmax, v);
      }
      #pragma unroll
      for (int off = 1; off < 16; off <<= 1)
        vmax = fmaxf(vmax, __shfl_xor(vmax, off, 32));
      const float mnew = fmaxf(m[r], vmax);
      const float esc  = __expf(m[r] - mnew);
      float rs = 0.f;
      #pragma unroll
      for (int nt = 0; nt < 4; ++nt) {
        const float p = __expf(s4[nt][r] - mnew);
        Ps[wave * 1024 + (half * 8 + r) * 64 + nt * 16 + lp] = (bf16_t)p;
        rs += p;
      }
      #pragma unroll
      for (int off = 1; off < 16; off <<= 1)
        rs += __shfl_xor(rs, off, 32);
      l[r] = l[r] * esc + rs;
      m[r] = mnew;
      #pragma unroll
      for (int dt = 0; dt < 8; ++dt) o[dt][r] *= esc;
    }

    // ---- O += P @ V : 2 K-steps x 8 head-dim tiles = 16 WMMAs ----
    #pragma unroll
    for (int ks2 = 0; ks2 < 2; ++ks2) {
      ABFrag pf;
      #pragma unroll
      for (int j = 0; j < 8; ++j) {
        const int k = ks2 * 32 + half * 16 + 2 * j;
        pf.u[j] = Pu[(wave * 1024 + lp * 64 + k) >> 1];
      }
      #pragma unroll
      for (int dt = 0; dt < 8; ++dt) {
        ABFrag vf;
        const int col = dt * 16 + lp;      // head-dim column
        #pragma unroll
        for (int j = 0; j < 8; ++j) {
          const int k = ks2 * 32 + half * 16 + 2 * j;  // key index
          vf.u[j] = Vu[(col * 64 + k) >> 1];
        }
        o[dt] = wmma_bf16(pf, vf, o[dt]);
      }
    }
    __syncthreads();
  }

  // ---- normalize and store bf16 attention output [2048 x 4096] ----
  #pragma unroll
  for (int dt = 0; dt < 8; ++dt)
    #pragma unroll
    for (int r = 0; r < 8; ++r) {
      const int qr = qbase + wave * 16 + half * 8 + r;
      const float val = o[dt][r] / l[r];
      Ob[(size_t)qr * 4096 + head * 128 + dt * 16 + lp] = (bf16_t)val;
    }
}

// ---------------------------------------------------------------------------
extern "C" void kernel_launch(void* const* d_in, const int* in_sizes, int n_in,
                              void* d_out, int out_size, void* d_ws, size_t ws_size,
                              hipStream_t stream)
{
  const float* x    = (const float*)d_in[0];
  // d_in[1] = mask: equivalent causal mask computed inline in attn_kernel
  const float* fc   = (const float*)d_in[2];
  const float* fs   = (const float*)d_in[3];
  const float* wqkv = (const float*)d_in[4];
  const float* wo   = (const float*)d_in[5];
  float* out        = (float*)d_out;

  char* ws = (char*)d_ws;
  bf16_t* qkvb = (bf16_t*)ws;                                 // 2048*6144*2 = 24 MB
  bf16_t* Ob   = (bf16_t*)(ws + (size_t)32 * 1024 * 1024);    // 2048*4096*2 = 16 MB

  // 1) qkv = x @ wqkv^T  (f32 in -> bf16 out)
  gemm_bt_kernel<false, true><<<dim3(6144 / 128, 2048 / 128), 256, 0, stream>>>(
      (const void*)x, wqkv, (void*)qkvb, 6144, 4096);

  // 2) RoPE on Q and K slices, in place
  rope_kernel<<<(2048 * 2560) / 256, 256, 0, stream>>>(qkvb, fc, fs);

  // 3) flash attention -> Ob (bf16)
  attn_kernel<<<dim3(16, 32), 256, 0, stream>>>(qkvb, Ob);

  // 4) out = Ob @ wo^T  (bf16 A, f32 out)
  gemm_bt_kernel<true, false><<<dim3(4096 / 128, 2048 / 128), 256, 0, stream>>>(
      (const void*)Ob, wo, (void*)out, 4096, 4096);
}
